// LSTMTransformer_45810121179466
// MI455X (gfx1250) — matrix operand order repaired and checked by reading, not compiled, not run on hardware
//
#include <hip/hip_runtime.h>
#include <hip/hip_bf16.h>
#include <stdint.h>

// LSTM: H=1024, T=4096, 2 layers, batch=1. fp32 in/out, bf16 WMMA for the
// big input GEMMs, fp32 persistent-grid scan for the recurrence.

#define HDIM   1024
#define TDIM   4096
#define FOURH  4096
#define NBLK_SCAN 64

typedef __attribute__((ext_vector_type(16))) __bf16 v16bf;
typedef __attribute__((ext_vector_type(8)))  float  v8f;

__device__ __forceinline__ float sigmoidf_(float x) {
  return 1.0f / (1.0f + __expf(-x));
}

// ---------------------------------------------------------------------------
// Grid-wide barrier (monotonic generation counter).
// bar[0] = arrival count, bar[1] = generation.
// ---------------------------------------------------------------------------
__device__ __forceinline__ void grid_barrier(unsigned* bar) {
  __syncthreads();
  if (threadIdx.x == 0) {
    __threadfence();
    unsigned g = __hip_atomic_load(&bar[1], __ATOMIC_RELAXED, __HIP_MEMORY_SCOPE_AGENT);
    unsigned a = __hip_atomic_fetch_add(&bar[0], 1u, __ATOMIC_ACQ_REL, __HIP_MEMORY_SCOPE_AGENT);
    if (a == NBLK_SCAN - 1) {
      __hip_atomic_store(&bar[0], 0u, __ATOMIC_RELAXED, __HIP_MEMORY_SCOPE_AGENT);
      __hip_atomic_fetch_add(&bar[1], 1u, __ATOMIC_RELEASE, __HIP_MEMORY_SCOPE_AGENT);
    } else {
      while (__hip_atomic_load(&bar[1], __ATOMIC_ACQUIRE, __HIP_MEMORY_SCOPE_AGENT) == g) {
        __builtin_amdgcn_s_sleep(2);
      }
    }
    __threadfence();
  }
  __syncthreads();
}

// ---------------------------------------------------------------------------
// Small helpers
// ---------------------------------------------------------------------------
__global__ void cvt_bf16(const float* __restrict__ in, __bf16* __restrict__ out, int n) {
  int i = blockIdx.x * blockDim.x + threadIdx.x;
  if (i < n) out[i] = (__bf16)in[i];
}

__global__ void bias_add(const float* __restrict__ a, const float* __restrict__ b,
                         float* __restrict__ o, int n) {
  int i = blockIdx.x * blockDim.x + threadIdx.x;
  if (i < n) o[i] = a[i] + b[i];
}

__global__ void init_bar(unsigned* bar) {
  if (threadIdx.x == 0) { bar[0] = 0u; bar[1] = 0u; }
}

// ---------------------------------------------------------------------------
// WMMA GEMM:  gx[T,4H] = A[T,H](bf16) * W[4H,H](bf16)^T + bias[4H]
//
// Block = 8 waves (256 thr). Block tile = 16 rows x 256 cols.
// The full A row-slab (16 x 1024 bf16 = 32 KB) is staged in LDS ONCE; the
// K loop then runs with no barriers: 2 ds_load_b128 (A frag, shared) +
// 4 global_load_b128 (two B frags) + 2 v_wmma_f32_16x16x32_bf16 per step,
// each A fragment feeding two C tiles (cols wave*32 and wave*32+16).
// A and B fragments share contiguous k order k = 16*half + s (wave32).
// ---------------------------------------------------------------------------
__global__ __launch_bounds__(256)
void wmma_gemm_gx(const __bf16* __restrict__ A, const __bf16* __restrict__ W,
                  const float* __restrict__ bias, float* __restrict__ gx) {
  __shared__ __bf16 As[16 * HDIM];   // 32 KB
  const int lane = threadIdx.x & 31;
  const int wave = threadIdx.x >> 5;
  const int half = lane >> 4;        // 0 or 1
  const int l16  = lane & 15;
  const int t0   = blockIdx.y * 16;

  // Stage A slab [16 x 1024] once: 2048 uint4, 8 per thread, coalesced b128s.
  {
    const uint4* src = (const uint4*)(A + (size_t)t0 * HDIM);
    uint4* dst = (uint4*)As;
#pragma unroll
    for (int i = 0; i < 8; ++i)
      dst[threadIdx.x + 256 * i] = src[threadIdx.x + 256 * i];
  }
  __syncthreads();

  const int col0 = blockIdx.x * 256 + wave * 32 + l16;  // two C tiles per wave
  const int col1 = col0 + 16;
  const __bf16* wrow0 = W + (size_t)col0 * HDIM;
  const __bf16* wrow1 = W + (size_t)col1 * HDIM;

  v8f acc0 = {};
  v8f acc1 = {};

#pragma unroll 4
  for (int k0 = 0; k0 < HDIM; k0 += 32) {
    // A fragment (16x32 bf16): lane row = l16; K = half*16 .. +15 contiguous
    union { v16bf v; uint4 q[2]; } af;
    const __bf16* ap = &As[l16 * HDIM + k0 + half * 16];
    af.q[0] = *(const uint4*)(ap);
    af.q[1] = *(const uint4*)(ap + 8);

    // B fragments (32x16 bf16): lane col = l16; K = half*16 .. +15 contiguous
    union { v16bf v; uint4 q[2]; } b0, b1;
    const __bf16* wp0 = wrow0 + k0 + half * 16;
    const __bf16* wp1 = wrow1 + k0 + half * 16;
    b0.q[0] = *(const uint4*)(wp0);
    b0.q[1] = *(const uint4*)(wp0 + 8);
    b1.q[0] = *(const uint4*)(wp1);
    b1.q[1] = *(const uint4*)(wp1 + 8);
    if (k0 + 32 < HDIM) {
      __builtin_prefetch(wp0 + 32, 0, 1);   // global_prefetch_b8
      __builtin_prefetch(wp1 + 32, 0, 1);
    }

    acc0 = __builtin_amdgcn_wmma_f32_16x16x32_bf16(
        false, af.v, false, b0.v, (short)0, acc0, false, false);
    acc1 = __builtin_amdgcn_wmma_f32_16x16x32_bf16(
        false, af.v, false, b1.v, (short)0, acc1, false, false);
  }

  // C/D layout: VGPR r -> row t0 + r + 8*half, col = l16 slot
  const float bb0 = bias[col0];
  const float bb1 = bias[col1];
#pragma unroll
  for (int r = 0; r < 8; ++r) {
    const int row = t0 + r + half * 8;
    gx[(size_t)row * FOURH + col0] = acc0[r] + bb0;
    gx[(size_t)row * FOURH + col1] = acc1[r] + bb1;
  }
}

// ---------------------------------------------------------------------------
// Persistent recurrent scan. 64 blocks x 256 threads. Block b owns hidden
// units [16b, 16b+16): computes all 4 gate dot-products (64 dots = 8 waves x 8),
// updates c (LDS-resident) and h, then grid-barriers. h double-buffered.
// w_hh stays fp32 (L2-resident, 16 MB; 192 MB L2 -> no HBM traffic per step).
// ---------------------------------------------------------------------------
__global__ __launch_bounds__(256)
void lstm_scan(const float* __restrict__ gx,   // [T,4H]
               const float* __restrict__ whh,  // [4H,H]
               float* __restrict__ hbuf,       // [2*H] ping-pong
               float* __restrict__ hout,       // [T,H] fp32 or nullptr
               __bf16* __restrict__ hout_bf,   // [T,H] bf16 or nullptr
               unsigned* __restrict__ bar) {
  __shared__ float gds[64];
  __shared__ float c_s[16];
  const int lane = threadIdx.x & 31;
  const int wave = threadIdx.x >> 5;
  const int j0   = blockIdx.x * 16;

  if (threadIdx.x < 16) {
    c_s[threadIdx.x] = 0.0f;
    hbuf[j0 + threadIdx.x] = 0.0f;   // zero the t=0 read buffer
  }
  grid_barrier(bar);

  for (int t = 0; t < TDIM; ++t) {
    const float* hin   = hbuf + (t & 1) * HDIM;
    float*       hnext = hbuf + ((t + 1) & 1) * HDIM;

    // Each lane caches h[4*lane + 128*m + c] (32 values) in registers.
    float4 hreg[8];
    const float4* hv = (const float4*)hin;
#pragma unroll
    for (int m = 0; m < 8; ++m) hreg[m] = hv[lane + 32 * m];

    // 8 dots per wave: d = wave*8+s;  gate = d>>4, unit = d&15
#pragma unroll
    for (int s = 0; s < 8; ++s) {
      const int d    = wave * 8 + s;
      const int gate = d >> 4;
      const int ul   = d & 15;
      const int n    = gate * HDIM + j0 + ul;  // row in [4H,H]
      const float4* wp = (const float4*)(whh + (size_t)n * HDIM) + lane;
      float sum = 0.0f;
#pragma unroll
      for (int m = 0; m < 8; ++m) {
        float4 w4 = wp[32 * m];
        sum = __builtin_fmaf(w4.x, hreg[m].x, sum);
        sum = __builtin_fmaf(w4.y, hreg[m].y, sum);
        sum = __builtin_fmaf(w4.z, hreg[m].z, sum);
        sum = __builtin_fmaf(w4.w, hreg[m].w, sum);
      }
#pragma unroll
      for (int off = 16; off > 0; off >>= 1)
        sum += __shfl_xor(sum, off, 32);     // wave32 reduction
      if (lane == 0) gds[d] = sum + gx[(size_t)t * FOURH + n];
    }
    __syncthreads();

    if (threadIdx.x < 16) {
      const int ul = threadIdx.x;
      const int j  = j0 + ul;
      const float iv = sigmoidf_(gds[ul]);
      const float fv = sigmoidf_(gds[16 + ul]);
      const float gv = tanhf(gds[32 + ul]);
      const float ov = sigmoidf_(gds[48 + ul]);
      const float c  = fv * c_s[ul] + iv * gv;
      c_s[ul] = c;
      const float h = ov * tanhf(c);
      hnext[j] = h;
      if (hout)    hout[(size_t)t * HDIM + j] = h;
      if (hout_bf) hout_bf[(size_t)t * HDIM + j] = (__bf16)h;
    }
    grid_barrier(bar);
  }
}

// ---------------------------------------------------------------------------
// Launch
// ---------------------------------------------------------------------------
extern "C" void kernel_launch(void* const* d_in, const int* in_sizes, int n_in,
                              void* d_out, int out_size, void* d_ws, size_t ws_size,
                              hipStream_t stream) {
  (void)in_sizes; (void)n_in; (void)out_size; (void)ws_size;
  const float* x     = (const float*)d_in[0];
  const float* w_ih0 = (const float*)d_in[1];
  const float* w_hh0 = (const float*)d_in[2];
  const float* b_ih0 = (const float*)d_in[3];
  const float* b_hh0 = (const float*)d_in[4];
  const float* w_ih1 = (const float*)d_in[5];
  const float* w_hh1 = (const float*)d_in[6];
  const float* b_ih1 = (const float*)d_in[7];
  const float* b_hh1 = (const float*)d_in[8];
  float* out = (float*)d_out;

  // Workspace carve-out (~96 MB total)
  char* ws = (char*)d_ws;
  size_t off = 0;
  auto carve = [&](size_t bytes) -> char* {
    char* p = ws + off;
    off = (off + bytes + 255) & ~(size_t)255;
    return p;
  };
  unsigned* bar   = (unsigned*)carve(256);
  float*    bias0 = (float*)carve((size_t)FOURH * 4);
  float*    bias1 = (float*)carve((size_t)FOURH * 4);
  float*    hbuf  = (float*)carve((size_t)2 * HDIM * 4);
  __bf16*   Xb    = (__bf16*)carve((size_t)TDIM * HDIM * 2);
  __bf16*   Wb0   = (__bf16*)carve((size_t)FOURH * HDIM * 2);
  __bf16*   Wb1   = (__bf16*)carve((size_t)FOURH * HDIM * 2);
  __bf16*   H1b   = (__bf16*)carve((size_t)TDIM * HDIM * 2);
  float*    gxbuf = (float*)carve((size_t)TDIM * FOURH * 4);

  const int NE = TDIM * HDIM;    // x / h elements
  const int NW = FOURH * HDIM;   // w_ih elements

  cvt_bf16<<<(NE + 255) / 256, 256, 0, stream>>>(x, Xb, NE);
  cvt_bf16<<<(NW + 255) / 256, 256, 0, stream>>>(w_ih0, Wb0, NW);
  cvt_bf16<<<(NW + 255) / 256, 256, 0, stream>>>(w_ih1, Wb1, NW);
  bias_add<<<(FOURH + 255) / 256, 256, 0, stream>>>(b_ih0, b_hh0, bias0, FOURH);
  bias_add<<<(FOURH + 255) / 256, 256, 0, stream>>>(b_ih1, b_hh1, bias1, FOURH);
  init_bar<<<1, 32, 0, stream>>>(bar);

  dim3 ggrid(FOURH / 256, TDIM / 16);   // 16 x 256 blocks

  // Layer 0
  wmma_gemm_gx<<<ggrid, 256, 0, stream>>>(Xb, Wb0, bias0, gxbuf);
  lstm_scan<<<NBLK_SCAN, 256, 0, stream>>>(gxbuf, w_hh0, hbuf, nullptr, H1b, bar);
  // Layer 1
  wmma_gemm_gx<<<ggrid, 256, 0, stream>>>(H1b, Wb1, bias1, gxbuf);
  lstm_scan<<<NBLK_SCAN, 256, 0, stream>>>(gxbuf, w_hh1, hbuf, out, nullptr, bar);
}
